// DotProductAttention_79929341378723
// MI455X (gfx1250) — matrix operand order repaired and checked by reading, not compiled, not run on hardware
//
#include <hip/hip_runtime.h>
#include <hip/hip_bf16.h>
#include <math.h>

typedef __attribute__((ext_vector_type(2)))  float    v2f;
typedef __attribute__((ext_vector_type(8)))  float    v8f;
typedef __attribute__((ext_vector_type(8)))  _Float16 v8h;
typedef __attribute__((ext_vector_type(16))) _Float16 v16h;

#define B_DIM 8
#define LQ 4096
#define LV 4096
#define DH 128   // D_HID == D_OUT == 128
#define NT (LV / 32)

__device__ __forceinline__ float gelu_exact(float x) {
    return 0.5f * x * (1.0f + erff(x * 0.70710678118654752f));
}

__device__ __forceinline__ v8f wmma_f32(v2f a, v2f b, v8f c) {
    return __builtin_amdgcn_wmma_f32_16x16x4_f32(false, a, false, b, (short)0, c, false, false);
}
__device__ __forceinline__ v8f wmma_f16(v16h a, v16h b, v8f c) {
    return __builtin_amdgcn_wmma_f32_16x16x32_f16(false, a, false, b, (short)0, c, false, false);
}
__device__ __forceinline__ v16h cat16(v8h a, v8h b) {
    return __builtin_shufflevector(a, b, 0,1,2,3,4,5,6,7,8,9,10,11,12,13,14,15);
}

// Async global->LDS 16B copy (ASYNCcnt-tracked, GVS mode: SGPR64 base + VGPR32 off)
__device__ __forceinline__ void async_cp16(unsigned lds_addr, const void* sbase, unsigned voff) {
    asm volatile("global_load_async_to_lds_b128 %0, %1, %2 offset:0"
                 :: "v"(lds_addr), "v"(voff), "s"(sbase) : "memory");
}
// ASYNCcnt decrements in issue order: <=4 waits for the previous tile only,
// leaving the 4 just-issued (next-tile) copies in flight behind the compute.
__device__ __forceinline__ void wait_async4() {
    asm volatile("s_wait_asynccnt 4" ::: "memory");
}
__device__ __forceinline__ void wait_async0() {
    asm volatile("s_wait_asynccnt 0" ::: "memory");
}

// ---------------------------------------------------------------------------
// Y = (GELU_OUT ? gelu : id)(X @ W + bias), X:[N,128] f32, Y:[N,128] OUT_T.
// Full f32 precision via V_WMMA_F32_16X16X4_F32 (tiny fraction of total work).
// ---------------------------------------------------------------------------
template<bool GELU_OUT, typename OUT_T>
__global__ __launch_bounds__(128) void linear128(const float* __restrict__ X,
                                                 const float* __restrict__ W,
                                                 const float* __restrict__ bias,
                                                 OUT_T* __restrict__ Y) {
    const int t    = threadIdx.x;
    const int w    = t >> 5;
    const int lane = t & 31;
    const int hi   = lane >> 4;
    const int lo   = lane & 15;

    const size_t row0 = (size_t)blockIdx.x * 64 + (size_t)w * 16;

    v2f a[32];
    const float* xrow = X + (row0 + (size_t)lo) * DH + 2 * hi;
#pragma unroll
    for (int kk = 0; kk < 32; ++kk)
        a[kk] = *(const v2f*)(xrow + 4 * kk);

#pragma unroll
    for (int n = 0; n < 8; ++n) {
        v8f acc = {};
#pragma unroll
        for (int kk = 0; kk < 32; ++kk) {
            const int kr = 4 * kk + 2 * hi;
            v2f b;
            b.x = W[(size_t)kr * DH + 16 * n + lo];
            b.y = W[(size_t)(kr + 1) * DH + 16 * n + lo];
            acc = wmma_f32(a[kk], b, acc);
        }
        const float bb = bias[16 * n + lo];
#pragma unroll
        for (int r = 0; r < 8; ++r) {
            float y = acc[r] + bb;
            if (GELU_OUT) y = gelu_exact(y);
            Y[(row0 + r + 8 * hi) * DH + 16 * n + lo] = (OUT_T)y;
        }
    }
}

// ---------------------------------------------------------------------------
// Fused attention, f16 WMMA (16x16x32), f32 accumulate & f32 softmax.
// WG = 4 waves = 64 q rows of one batch; 32-key tiles, double-buffered in LDS
// via global_load_async_to_lds_b128 (tile kt+1 streams behind compute of kt).
//   Pass A: online per-lane-slot max/sum (elementwise), shfl_xor combine.
//   Pass B: recompute scores (v is L2-resident; cheaper than a 1 GB raw score
//           round-trip), write normalized attn once, LDS->LDS transpose of v,
//           ctx += P(16x32) @ V(32x128) with one K=32 WMMA per h-tile.
// ---------------------------------------------------------------------------
__global__ __launch_bounds__(128) void attn_kernel(const _Float16* __restrict__ q16,
                                                   const _Float16* __restrict__ v16,
                                                   float* __restrict__ attn,
                                                   float* __restrict__ ctxg) {
    __shared__ _Float16 q_lds[64 * DH];      // 16 KB
    __shared__ _Float16 v_lds[2][32 * DH];   // 16 KB (double-buffered, row-major)
    __shared__ _Float16 vT_lds[DH * 32];     //  8 KB (transposed: h x key)
    __shared__ _Float16 p_lds[4][16 * 32];   //  4 KB (per-wave P tile)

    const int t    = threadIdx.x;
    const int w    = t >> 5;
    const int lane = t & 31;
    const int hi   = lane >> 4;
    const int lo   = lane & 15;

    const int b    = blockIdx.x >> 6;
    const int qblk = blockIdx.x & 63;

    const _Float16* qb = q16 + ((size_t)b * LQ + (size_t)qblk * 64) * DH;
    const _Float16* vb = v16 + (size_t)b * LV * DH;
    float* attn_b = attn + ((size_t)b * LQ + (size_t)qblk * 64 + (size_t)w * 16) * (size_t)LV;
    float* ctx_b  = ctxg + ((size_t)b * LQ + (size_t)qblk * 64 + (size_t)w * 16) * DH;

    // Stage the 64x128 f16 q block (1024 x 16B chunks)
    {
        v8h* qd = (v8h*)q_lds;
        const v8h* qs = (const v8h*)qb;
#pragma unroll
        for (int i = 0; i < 8; ++i) qd[i * 128 + t] = qs[i * 128 + t];
    }
    __syncthreads();

    const int qrow = w * 16;
    // q A-fragments are loop-invariant: hoist (4 x v16h = 32 VGPRs)
    v16h aq[4];
#pragma unroll
    for (int kk = 0; kk < 4; ++kk) {
        const int kb = 32 * kk;
        aq[kk] = cat16(*(const v8h*)&q_lds[(qrow + lo) * DH + kb + 16 * hi],
                       *(const v8h*)&q_lds[(qrow + lo) * DH + kb + 16 * hi + 8]);
    }

    const unsigned vbase[2] = { (unsigned)(size_t)&v_lds[0][0],
                                (unsigned)(size_t)&v_lds[1][0] };
    auto issue_tile = [&](int kt) {
        const void* src = (const void*)(vb + (size_t)kt * 32 * DH);
        const unsigned base = vbase[kt & 1];
#pragma unroll
        for (int i = 0; i < 4; ++i) {
            const unsigned c = (unsigned)(i * 128 + t);   // 16B chunk index
            async_cp16(base + c * 16u, src, c * 16u);
        }
    };

    // ---------------- Pass A: online max / sum(exp) -----------------------
    float mloc[8], sloc[8];
#pragma unroll
    for (int r = 0; r < 8; ++r) { mloc[r] = -3.0e38f; sloc[r] = 0.0f; }

    issue_tile(0);
    for (int kt = 0; kt < NT; ++kt) {
        if (kt + 1 < NT) { issue_tile(kt + 1); wait_async4(); }
        else             { wait_async0(); }
        __syncthreads();                      // tile kt landed (all waves)
        const _Float16* vcur = v_lds[kt & 1];

        v8f S0 = {}, S1 = {};
#pragma unroll
        for (int kk = 0; kk < 4; ++kk) {
            const int kb = 32 * kk;
            v16h b0 = cat16(*(const v8h*)&vcur[lo * DH + kb + 16 * hi],
                            *(const v8h*)&vcur[lo * DH + kb + 16 * hi + 8]);
            v16h b1 = cat16(*(const v8h*)&vcur[(16 + lo) * DH + kb + 16 * hi],
                            *(const v8h*)&vcur[(16 + lo) * DH + kb + 16 * hi + 8]);
            S0 = wmma_f16(aq[kk], b0, S0);
            S1 = wmma_f16(aq[kk], b1, S1);
        }
#pragma unroll
        for (int r = 0; r < 8; ++r) {
            const float s0 = S0[r], s1 = S1[r];
            const float mn = fmaxf(mloc[r], fmaxf(s0, s1));
            sloc[r] = sloc[r] * __expf(mloc[r] - mn) + __expf(s0 - mn) + __expf(s1 - mn);
            mloc[r] = mn;
        }
        __syncthreads();                      // all reads done before buffer reuse
    }

    // Combine the 16 column slots of each row (within 16-lane groups)
#pragma unroll
    for (int off = 8; off >= 1; off >>= 1) {
#pragma unroll
        for (int r = 0; r < 8; ++r) {
            const float mo = __shfl_xor(mloc[r], off, 16);
            const float so = __shfl_xor(sloc[r], off, 16);
            const float mn = fmaxf(mloc[r], mo);
            sloc[r] = sloc[r] * __expf(mloc[r] - mn) + so * __expf(mo - mn);
            mloc[r] = mn;
        }
    }
    float inv[8];
#pragma unroll
    for (int r = 0; r < 8; ++r) inv[r] = 1.0f / sloc[r];

    // ---------------- Pass B: attn out + ctx accumulate -------------------
    v8f ctxa[8];
    {
        v8f z = {};
#pragma unroll
        for (int h = 0; h < 8; ++h) ctxa[h] = z;
    }

    issue_tile(0);
    for (int kt = 0; kt < NT; ++kt) {
        if (kt + 1 < NT) { issue_tile(kt + 1); wait_async4(); }
        else             { wait_async0(); }
        __syncthreads();                      // tile kt landed (all waves)
        const _Float16* vcur = v_lds[kt & 1];

        // scores (reads row-major v only)
        v8f S0 = {}, S1 = {};
#pragma unroll
        for (int kk = 0; kk < 4; ++kk) {
            const int kb = 32 * kk;
            v16h b0 = cat16(*(const v8h*)&vcur[lo * DH + kb + 16 * hi],
                            *(const v8h*)&vcur[lo * DH + kb + 16 * hi + 8]);
            v16h b1 = cat16(*(const v8h*)&vcur[(16 + lo) * DH + kb + 16 * hi],
                            *(const v8h*)&vcur[(16 + lo) * DH + kb + 16 * hi + 8]);
            S0 = wmma_f16(aq[kk], b0, S0);
            S1 = wmma_f16(aq[kk], b1, S1);
        }

        // normalized probabilities; write attn (once) + per-wave P tile (f16)
#pragma unroll
        for (int r = 0; r < 8; ++r) {
            const float p0 = __expf(S0[r] - mloc[r]) * inv[r];
            const float p1 = __expf(S1[r] - mloc[r]) * inv[r];
            const size_t row = (size_t)(r + 8 * hi);
            attn_b[row * LV + kt * 32 + lo]      = p0;
            attn_b[row * LV + kt * 32 + 16 + lo] = p1;
            p_lds[w][(r + 8 * hi) * 32 + lo]      = (_Float16)p0;
            p_lds[w][(r + 8 * hi) * 32 + 16 + lo] = (_Float16)p1;
        }

        // LDS->LDS transpose of tile kt (vT last read before this barrier pair)
#pragma unroll
        for (int i = 0; i < 4; ++i) {
            const int c = i * 128 + t;
            v8h val = *(const v8h*)(vcur + c * 8);
            const int row = c >> 4, col0 = (c & 15) * 8;
#pragma unroll
            for (int e = 0; e < 8; ++e)
                vT_lds[(col0 + e) * 32 + row] = val[e];
        }
        __syncthreads();                      // vT + P visible

        // ctx += P(16x32) @ V(32x128): reads vT/p_lds only -> the row-major
        // buffer can be overwritten by issue(kt+2) right after this iteration.
        v16h ap = cat16(*(const v8h*)&p_lds[w][lo * 32 + 16 * hi],
                        *(const v8h*)&p_lds[w][lo * 32 + 16 * hi + 8]);
#pragma unroll
        for (int h = 0; h < 8; ++h) {
            v16h bv = cat16(*(const v8h*)&vT_lds[(16 * h + lo) * 32 + 16 * hi],
                            *(const v8h*)&vT_lds[(16 * h + lo) * 32 + 16 * hi + 8]);
            ctxa[h] = wmma_f16(ap, bv, ctxa[h]);
        }
    }

    // Epilogue: gelu(ctx) -> f32 workspace for the final (f32 WMMA) linear
#pragma unroll
    for (int h = 0; h < 8; ++h) {
#pragma unroll
        for (int r = 0; r < 8; ++r) {
            ctx_b[(size_t)(r + 8 * hi) * DH + 16 * h + lo] = gelu_exact(ctxa[h][r]);
        }
    }
}

// ---------------------------------------------------------------------------
extern "C" void kernel_launch(void* const* d_in, const int* in_sizes, int n_in,
                              void* d_out, int out_size, void* d_ws, size_t ws_size,
                              hipStream_t stream) {
    const float* query = (const float*)d_in[0];
    const float* value = (const float*)d_in[1];
    const float* Wq    = (const float*)d_in[2];
    const float* bq    = (const float*)d_in[3];
    const float* Wv    = (const float*)d_in[4];
    const float* bv    = (const float*)d_in[5];
    const float* Wo    = (const float*)d_in[6];
    const float* bo    = (const float*)d_in[7];

    float* out_ptr  = (float*)d_out;
    float* attn_ptr = out_ptr + (size_t)B_DIM * LQ * DH;

    const size_t act = (size_t)B_DIM * LQ * DH;     // 4,194,304 elements
    _Float16* q16 = (_Float16*)d_ws;                //  8 MB
    _Float16* v16 = q16 + act;                      //  8 MB
    float*   ctxg = (float*)(v16 + act);            // 16 MB  (32 MB ws total)

    const int nblk = (B_DIM * LQ) / 64;             // 512 blocks, 128 threads

    linear128<true,  _Float16><<<nblk, 128, 0, stream>>>(query, Wq, bq, q16);
    linear128<true,  _Float16><<<nblk, 128, 0, stream>>>(value, Wv, bv, v16);
    attn_kernel<<<nblk, 128, 0, stream>>>(q16, v16, attn_ptr, ctxg);
    linear128<false, float   ><<<nblk, 128, 0, stream>>>(ctxg, Wo, bo, out_ptr);
}
